// Seq2Seq_62019327754324
// MI455X (gfx1250) — matrix operand order repaired and checked by reading, compiled, not run on hardware
//
#include <hip/hip_runtime.h>

// ---------------- problem constants ----------------
#define E_DIM    512
#define B_DIM    2048
#define T_DIM    100
#define F_DIM    10
#define POSE_DIM 34
#define G_DIM    (4 * E_DIM)   // 2048 gate columns
#define K_DIM    (2 * E_DIM)   // 1024 fused K ([x | h])

typedef __attribute__((ext_vector_type(16))) __bf16 v16bf;
typedef __attribute__((ext_vector_type(8)))  float  v8f;

__device__ __forceinline__ float sigmoid_f(float x) { return 1.0f / (1.0f + __expf(-x)); }
// stable tanh: 1 - 2/(e^{2x}+1)  (correct limits at +/-inf)
__device__ __forceinline__ float tanh_f(float x)    { return 1.0f - 2.0f / (__expf(2.0f * x) + 1.0f); }

// ---------------- weight packing: Bw[k, n] = (k<E ? Wih[n,k] : Whh[n,k-E]) as bf16 ----------------
__global__ void pack_weights(const float* __restrict__ Wih, const float* __restrict__ Whh,
                             const float* __restrict__ bih, const float* __restrict__ bhh,
                             __bf16* __restrict__ Bw, float* __restrict__ bias)
{
    int i = blockIdx.x * blockDim.x + threadIdx.x;
    if (i < G_DIM) bias[i] = bih[i] + bhh[i];
    if (i >= K_DIM * G_DIM) return;
    int k = i >> 11;          // / G_DIM
    int n = i & (G_DIM - 1);
    float w = (k < E_DIM) ? Wih[(size_t)n * E_DIM + k]
                          : Whh[(size_t)n * E_DIM + (k - E_DIM)];
    Bw[i] = (__bf16)w;
}

// ---------------- input projection for ONE timestep (fallback path) ----------------
__global__ void encode_step(const float* __restrict__ x,     // [B, T, POSE]
                            const float* __restrict__ W,     // [E, POSE]
                            const float* __restrict__ b,     // [E]
                            int t, __bf16* __restrict__ xin) // [B, E]
{
    int i = blockIdx.x * blockDim.x + threadIdx.x;
    if (i >= B_DIM * E_DIM) return;
    int bi = i >> 9;
    int e  = i & (E_DIM - 1);
    const float* xp = x + ((size_t)bi * T_DIM + t) * POSE_DIM;
    const float* wp = W + (size_t)e * POSE_DIM;
    float s = b[e];
#pragma unroll
    for (int p = 0; p < POSE_DIM; ++p) s += xp[p] * wp[p];
    xin[i] = (__bf16)s;
}

// ---------------- input projection for ALL timesteps in one launch (preferred path) ----------------
__global__ void encode_all(const float* __restrict__ x,      // [B, T, POSE]
                           const float* __restrict__ W,      // [E, POSE]
                           const float* __restrict__ b,      // [E]
                           __bf16* __restrict__ xall)        // [T, B, E]
{
    long long i = (long long)blockIdx.x * blockDim.x + threadIdx.x;
    if (i >= (long long)T_DIM * B_DIM * E_DIM) return;
    int e  = (int)(i & (E_DIM - 1));
    int bi = (int)((i >> 9) & (B_DIM - 1));
    int t  = (int)(i >> 20);                 // / (B*E) = / 2^20
    const float* xp = x + ((size_t)bi * T_DIM + t) * POSE_DIM;
    const float* wp = W + (size_t)e * POSE_DIM;
    float s = b[e];
#pragma unroll
    for (int p = 0; p < POSE_DIM; ++p) s += xp[p] * wp[p];
    xall[i] = (__bf16)s;
}

// ---------------- fully fused LSTM cell step ----------------------------------------------------
// gates = [Xin | H] @ Bw + bias, then i/f/g/o -> (h, c) applied in-kernel via an LDS gate exchange.
// Workgroup: 128 rows x 32 e-columns, covering those columns of ALL FOUR gates.
// 8 waves = 2 row-halves x 4 gates; each wave: 4x2 WMMA accumulators (bf16 16x16x32, fp32 acc).
#define TM    128
#define TE    32    // e-columns per workgroup (per gate)
#define KC    64
#define LDA_S 72    // bf16 elems per A row in LDS (144 B, 16B aligned)
#define LDB_S 136   // bf16 elems per B row in LDS (272 B, 16B aligned)
#define LDG_S 132   // fp32 row stride of gate exchange: 32 e * 4 gates + 4 pad (528 B, 16B aligned)

union alignas(16) CellSmem {
    struct { __bf16 As[TM * LDA_S]; __bf16 Bs[KC * LDB_S]; } stage;
    float gbuf[TM * LDG_S];     // [row][e*4 + gate]  -> one b128 read yields all 4 gates
};

__global__ __launch_bounds__(256) void lstm_cell_fused(
    const __bf16* __restrict__ Xin,  // [B, E]  (K = 0..511)
    const __bf16* __restrict__ Hst,  // [B, E]  (K = 512..1023)
    const __bf16* __restrict__ Bw,   // [K_DIM, G_DIM] row-major bf16
    const float*  __restrict__ bias, // [G_DIM]
    float*        __restrict__ c,    // [B, E] in/out cell state
    __bf16*       __restrict__ h_bf, // [B, E] out hidden (next-step GEMM input)
    float*        __restrict__ preds)// optional [B, E] fp32 slot (decoder), else nullptr
{
    __shared__ CellSmem sm;

    const int tid  = threadIdx.x;
    const int lane = tid & 31;
    const int wv   = tid >> 5;   // 0..7
    const int wm   = wv & 1;     // row half (2)
    const int gi   = wv >> 1;    // gate index (4): i,f,g,o
    const int bm   = blockIdx.y * TM;       // batch-row tile
    const int be   = blockIdx.x * TE;       // e-column tile (within each gate)

    v8f acc[4][2];
#pragma unroll
    for (int i = 0; i < 4; ++i)
#pragma unroll
        for (int j = 0; j < 2; ++j)
#pragma unroll
            for (int r = 0; r < 8; ++r) acc[i][j][r] = 0.0f;

    for (int kk = 0; kk < K_DIM; kk += KC) {
        // ---- stage A tile (128 x 64 bf16): KC=64 aligns with the Xin/H split at k=512
        const __bf16* Asrc = (kk < E_DIM) ? Xin : Hst;
        const int     kb   = (kk < E_DIM) ? kk : (kk - E_DIM);
#pragma unroll
        for (int r = 0; r < 4; ++r) {
            int g    = tid + r * 256;        // 1024 groups of 8 bf16
            int row  = g >> 3;
            int col8 = (g & 7) << 3;
            *(uint4*)(&sm.stage.As[row * LDA_S + col8]) =
                *(const uint4*)(Asrc + (size_t)(bm + row) * E_DIM + kb + col8);
        }
        // ---- stage B tile (64 x 128 bf16): col c -> gate (c>>5), e-col (c&31)
#pragma unroll
        for (int r = 0; r < 4; ++r) {
            int g    = tid + r * 256;
            int row  = g >> 4;
            int col8 = (g & 15) << 3;                       // multiple of 8, stays in one gate strip
            int ncol = (col8 >> 5) * E_DIM + be + (col8 & 31);
            *(uint4*)(&sm.stage.Bs[row * LDB_S + col8]) =
                *(const uint4*)(Bw + (size_t)(kk + row) * G_DIM + ncol);
        }
        // hint the next K-chunk into cache while we compute
        if (kk + KC < K_DIM) {
            const __bf16* nsrc = (kk + KC < E_DIM) ? Xin : Hst;
            const int     nk   = (kk + KC < E_DIM) ? (kk + KC) : (kk + KC - E_DIM);
            __builtin_prefetch(nsrc + (size_t)(bm + (tid >> 3)) * E_DIM + nk + ((tid & 7) << 3), 0, 0);
            __builtin_prefetch(Bw + (size_t)(kk + KC + (tid >> 4)) * G_DIM +
                               ((((tid & 15) << 3) >> 5) * E_DIM + be + (((tid & 15) << 3) & 31)), 0, 0);
        }
        __syncthreads();

#pragma unroll
        for (int ks = 0; ks < KC; ks += 32) {
            // B fragments: lane%16 = N column, elements hold K {hs*8.., 16+hs*8..}, hs=lane/16 (same as A)
            v16bf bfrag[2];
#pragma unroll
            for (int j = 0; j < 2; ++j) {
                const __bf16* p = &sm.stage.Bs[(ks + (lane >> 4) * 8) * LDB_S + gi * TE + j * 16 + (lane & 15)];
#pragma unroll
                for (int s = 0; s < 8; ++s) {
                    bfrag[j][s]     = p[s * LDB_S];
                    bfrag[j][8 + s] = p[(16 + s) * LDB_S];
                }
            }
            // A fragments: lanes 0-15 row M=lane hold K {hs*8.., 16+hs*8..}, hs=lane/16
            const int hs  = lane >> 4;
            const int mrw = lane & 15;
#pragma unroll
            for (int i = 0; i < 4; ++i) {
                const __bf16* p = &sm.stage.As[(wm * 64 + i * 16 + mrw) * LDA_S + ks + hs * 8];
                v16bf afrag;
                ((uint4*)&afrag)[0] = *(const uint4*)(p);        // K = ks + hs*8 + 0..7
                ((uint4*)&afrag)[1] = *(const uint4*)(p + 16);   // K = ks + 16 + hs*8 + 0..7
#pragma unroll
                for (int j = 0; j < 2; ++j)
                    acc[i][j] = __builtin_amdgcn_wmma_f32_16x16x32_bf16(
                        false, afrag, false, bfrag[j], (short)0, acc[i][j], false, false);
            }
        }
        __syncthreads();
    }

    // ---- exchange gate values through LDS (C/D layout: VGPR r -> row r + 8*(lane/16), col lane%16)
    const int nl = lane & 15;
    const int mh = lane >> 4;
#pragma unroll
    for (int j = 0; j < 2; ++j) {
        int eloc = j * 16 + nl;                         // 0..31 within gate strip
        float bv = bias[gi * E_DIM + be + eloc];
#pragma unroll
        for (int i = 0; i < 4; ++i) {
            int row0 = wm * 64 + i * 16 + mh * 8;
#pragma unroll
            for (int r = 0; r < 8; ++r)
                sm.gbuf[(row0 + r) * LDG_S + eloc * 4 + gi] = acc[i][j][r] + bv;
        }
    }
    __syncthreads();

    // ---- elementwise LSTM update: 128 rows x 32 e = 4096 elems, 16 per thread
    const bool has_preds = (preds != nullptr);
#pragma unroll
    for (int r = 0; r < 16; ++r) {
        int idx = tid + r * 256;
        int row = idx >> 5;          // 0..127
        int e   = idx & 31;          // 0..31
        // one 16B LDS read returns {i, f, g, o} for this (row, e)
        const v8f* dummy = nullptr; (void)dummy;
        float4 gv = *(const float4*)(&sm.gbuf[row * LDG_S + e * 4]);
        float ig = sigmoid_f(gv.x);
        float fg = sigmoid_f(gv.y);
        float gg = tanh_f   (gv.z);
        float og = sigmoid_f(gv.w);
        size_t gidx = (size_t)(bm + row) * E_DIM + be + e;
        float cn = fg * c[gidx] + ig * gg;
        float hn = og * tanh_f(cn);
        c[gidx]    = cn;
        h_bf[gidx] = (__bf16)hn;
        if (has_preds) preds[gidx] = hn;
    }
}

// ---------------- output projection: out[b,f,p] = preds[f,b,:] . Wd[p,:] + bd[p] ----------------
__global__ void final_linear(const float* __restrict__ preds, // [F, B, E]
                             const float* __restrict__ Wd,    // [POSE, E]
                             const float* __restrict__ bd,    // [POSE]
                             float* __restrict__ out)         // [B, F, POSE]
{
    int i = blockIdx.x * blockDim.x + threadIdx.x;
    if (i >= B_DIM * F_DIM * POSE_DIM) return;
    int p = i % POSE_DIM;
    int f = (i / POSE_DIM) % F_DIM;
    int b = i / (POSE_DIM * F_DIM);
    const float* hp = preds + ((size_t)f * B_DIM + b) * E_DIM;
    const float* wp = Wd + (size_t)p * E_DIM;
    float s = bd[p];
#pragma unroll 8
    for (int e = 0; e < E_DIM; ++e) s += hp[e] * wp[e];
    out[i] = s;
}

// ---------------- host orchestration ----------------
extern "C" void kernel_launch(void* const* d_in, const int* in_sizes, int n_in,
                              void* d_out, int out_size, void* d_ws, size_t ws_size,
                              hipStream_t stream)
{
    const float* x     = (const float*)d_in[0];
    const float* encW  = (const float*)d_in[1];
    const float* encB  = (const float*)d_in[2];
    const float* Wih[3] = { (const float*)d_in[3], (const float*)d_in[7],  (const float*)d_in[11] };
    const float* Whh[3] = { (const float*)d_in[4], (const float*)d_in[8],  (const float*)d_in[12] };
    const float* bih[3] = { (const float*)d_in[5], (const float*)d_in[9],  (const float*)d_in[13] };
    const float* bhh[3] = { (const float*)d_in[6], (const float*)d_in[10], (const float*)d_in[14] };
    const float* decW  = (const float*)d_in[15];
    const float* decB  = (const float*)d_in[16];
    float* out = (float*)d_out;

    // workspace carve-up (256B aligned)
    char* base = (char*)d_ws;
    size_t off = 0;
    auto take = [&](size_t bytes) -> void* {
        void* p = base + off;
        off += (bytes + 255) & ~(size_t)255;
        return p;
    };
    __bf16* Bw[3]; float* bias[3];
    for (int ci = 0; ci < 3; ++ci) {
        Bw[ci]   = (__bf16*)take((size_t)K_DIM * G_DIM * sizeof(__bf16));
        bias[ci] = (float*) take((size_t)G_DIM * sizeof(float));
    }
    __bf16* xin  = (__bf16*)take((size_t)B_DIM * E_DIM * sizeof(__bf16));
    __bf16* h1bf = (__bf16*)take((size_t)B_DIM * E_DIM * sizeof(__bf16));
    __bf16* h2bf = (__bf16*)take((size_t)B_DIM * E_DIM * sizeof(__bf16));
    __bf16* zbf  = (__bf16*)take((size_t)B_DIM * E_DIM * sizeof(__bf16));
    float*  c1   = (float*) take((size_t)B_DIM * E_DIM * sizeof(float));
    float*  c2   = (float*) take((size_t)B_DIM * E_DIM * sizeof(float));
    float*  prds = (float*) take((size_t)F_DIM * B_DIM * E_DIM * sizeof(float));
    if (off > ws_size) return;  // workspace too small: bail deterministically

    // optional one-shot input-projection buffer for all T steps (removes 100 serial launches)
    const size_t xall_bytes = (size_t)T_DIM * B_DIM * E_DIM * sizeof(__bf16);
    __bf16* xall = nullptr;
    if (off + xall_bytes <= ws_size) xall = (__bf16*)take(xall_bytes);

    // initial state (every call: deterministic, no cached state)
    hipMemsetAsync(h1bf, 0, (size_t)B_DIM * E_DIM * sizeof(__bf16), stream);
    hipMemsetAsync(h2bf, 0, (size_t)B_DIM * E_DIM * sizeof(__bf16), stream);
    hipMemsetAsync(zbf,  0, (size_t)B_DIM * E_DIM * sizeof(__bf16), stream);
    hipMemsetAsync(c1,   0, (size_t)B_DIM * E_DIM * sizeof(float),  stream);
    hipMemsetAsync(c2,   0, (size_t)B_DIM * E_DIM * sizeof(float),  stream);

    // pack weights (bf16) + biases (fp32), each call
    {
        int n = K_DIM * G_DIM;
        int blocks = (n + 255) / 256;
        for (int ci = 0; ci < 3; ++ci)
            pack_weights<<<blocks, 256, 0, stream>>>(Wih[ci], Whh[ci], bih[ci], bhh[ci], Bw[ci], bias[ci]);
    }

    const dim3 ggrid(E_DIM / TE, B_DIM / TM);   // 16 e-tiles x 16 row-tiles
    const int  eblk = (B_DIM * E_DIM + 255) / 256;

    if (xall) {
        long long n = (long long)T_DIM * B_DIM * E_DIM;
        int blocks = (int)((n + 255) / 256);
        encode_all<<<blocks, 256, 0, stream>>>(x, encW, encB, xall);
    }

    // ---- encoder: 100 steps, 2 stacked cells, fully fused cell updates ----
    for (int t = 0; t < T_DIM; ++t) {
        const __bf16* x_t;
        if (xall) {
            x_t = xall + (size_t)t * B_DIM * E_DIM;
        } else {
            encode_step<<<eblk, 256, 0, stream>>>(x, encW, encB, t, xin);
            x_t = xin;
        }
        lstm_cell_fused<<<ggrid, 256, 0, stream>>>(x_t,  h1bf, Bw[0], bias[0], c1, h1bf, nullptr);
        lstm_cell_fused<<<ggrid, 256, 0, stream>>>(h1bf, h2bf, Bw[1], bias[1], c2, h2bf, nullptr);
    }

    // ---- decoder: carry (h2bf, c2); input is zeros at f=0, then previous h (== carried h) ----
    for (int f = 0; f < F_DIM; ++f) {
        const __bf16* xin_f = (f == 0) ? zbf : h2bf;
        lstm_cell_fused<<<ggrid, 256, 0, stream>>>(xin_f, h2bf, Bw[2], bias[2], c2, h2bf,
                                                   prds + (size_t)f * B_DIM * E_DIM);
    }

    // ---- output projection ----
    {
        int n = B_DIM * F_DIM * POSE_DIM;
        final_linear<<<(n + 255) / 256, 256, 0, stream>>>(prds, decW, decB, out);
    }
}
